// Critic_1881195675939
// MI455X (gfx1250) — matrix operand order repaired and checked by reading, not compiled, not run on hardware
//
#include <hip/hip_runtime.h>
#include <math.h>

typedef __attribute__((ext_vector_type(16))) _Float16 v16h;
typedef __attribute__((ext_vector_type(8)))  float    v8f;

#define NEG_SLOPE 0.2f
#define GAT_EPS 1e-16f

// ---------------- float atomic max (sign-aware int trick; m initialized to -inf) ---
__device__ inline void atomicMaxFloat(float* addr, float v) {
    if (v >= 0.0f) {
        atomicMax((int*)addr, __float_as_int(v));
    } else {
        atomicMin((unsigned int*)addr, __float_as_uint(v));
    }
}

__device__ inline float lrelu(float v) { return v > 0.0f ? v : NEG_SLOPE * v; }

__device__ inline void edge_sd(const long long* __restrict__ ei, long long E,
                               long long e, int& s, int& d) {
    if (e < E) { s = (int)ei[e]; d = (int)ei[E + e]; }
    else       { s = d = (int)(e - E); }
}

// ---------------- init: zero accumulators, -inf maxima, zero d_out ------------------
__global__ void init_ws_kernel(float* m1, float* s1, float* out1,
                               float* m2, float* s2, float* out2,
                               float* dout, int N) {
    long long i = (long long)blockIdx.x * blockDim.x + threadIdx.x;
    if (i == 0) dout[0] = 0.0f;
    if (i < 2LL * N) { m1[i] = -INFINITY; s1[i] = 0.0f; }
    if (i < (long long)N) { m2[i] = -INFINITY; s2[i] = 0.0f; out2[i] = 0.0f; }
    if (i < 32LL * N) out1[i] = 0.0f;
}

// ---------------- layer-1 GEMM via WMMA: xp[N,32] = x[N,128] @ W1[128,32] -----------
// W1 (16 KB f32) is pulled into LDS with the gfx1250 async-to-LDS pipe, then
// converted to a padded, transposed f16 tile. One wave computes 16 output rows
// with two 16x16 f32 accumulators; K=128 = 4 x v_wmma_f32_16x16x32_f16.
__global__ __launch_bounds__(256) void gemm1_wmma_kernel(const float* __restrict__ x,
                                                         const float* __restrict__ W1,
                                                         float* __restrict__ xp, int N) {
    __shared__ float    w_stage[128 * 32];   // raw f32 staging (16 KB)
    __shared__ _Float16 lds_w[32][136];      // [n][k] f16, padded rows

    const int tid = threadIdx.x;

    // --- async copy W1 -> LDS staging: 4 x b128 per lane, tracked with ASYNCcnt ----
    {
        const unsigned lbase = (unsigned)(uintptr_t)(&w_stage[0]);
        #pragma unroll
        for (int it = 0; it < 4; ++it) {
            const int fidx = (it * 256 + tid) * 4;                 // float index, 16B/lane
            const unsigned long long ga = (unsigned long long)(const void*)(W1 + fidx);
            const unsigned la = lbase + (unsigned)fidx * 4u;       // LDS byte offset
            asm volatile("global_load_async_to_lds_b128 %0, %1, off"
                         :: "v"(la), "v"(ga) : "memory");
        }
        asm volatile("s_wait_asynccnt 0x0" ::: "memory");
    }
    __syncthreads();

    // --- convert to f16, transposed [n][k] --------------------------------------
    for (int idx = tid; idx < 128 * 32; idx += 256) {
        const int k = idx >> 5, n = idx & 31;
        lds_w[n][k] = (_Float16)w_stage[idx];
    }
    __syncthreads();

    const int lane = tid & 31;
    const int wave = tid >> 5;
    const int row0 = blockIdx.x * 128 + wave * 16;
    if (row0 >= N) return;

    // A-matrix (16x32 f16) ISA layout: lanes 0-15 hold row M=lane, K in {0..7,16..23};
    // lanes 16-31 hold row M=lane-16, K in {8..15,24..31}.
    const int arow  = row0 + (lane & 15);
    const int arowc = arow < N ? arow : (N - 1);   // clamp loads; stores are guarded
    const float* xr = x + (size_t)arowc * 128;
    const int kA    = (lane >> 4) << 3;            // 0 or 8
    // B-matrix (32x16 f16): lanes 0-15 hold col N=lane, K in {0..7,16..23}; lanes 16-31: K in {8..15,24..31}.
    const int kB   = (lane >> 4) << 4;             // 0 or 16
    const int colB = lane & 15;

    v8f c0 = {}; v8f c1 = {};
    #pragma unroll
    for (int k0 = 0; k0 < 128; k0 += 32) {
        const float4 f0 = *(const float4*)(xr + k0 + kA);
        const float4 f1 = *(const float4*)(xr + k0 + kA + 4);
        const float4 f2 = *(const float4*)(xr + k0 + kA + 16);
        const float4 f3 = *(const float4*)(xr + k0 + kA + 20);
        v16h a;
        a[0]=(_Float16)f0.x;  a[1]=(_Float16)f0.y;  a[2]=(_Float16)f0.z;  a[3]=(_Float16)f0.w;
        a[4]=(_Float16)f1.x;  a[5]=(_Float16)f1.y;  a[6]=(_Float16)f1.z;  a[7]=(_Float16)f1.w;
        a[8]=(_Float16)f2.x;  a[9]=(_Float16)f2.y;  a[10]=(_Float16)f2.z; a[11]=(_Float16)f2.w;
        a[12]=(_Float16)f3.x; a[13]=(_Float16)f3.y; a[14]=(_Float16)f3.z; a[15]=(_Float16)f3.w;

        const _Float16* wp0 = &lds_w[colB][k0 + kA];
        const _Float16* wp1 = &lds_w[colB + 16][k0 + kA];
        v16h b0, b1;
        #pragma unroll
        for (int j = 0; j < 16; ++j) { b0[j] = wp0[j + (j & 8)]; b1[j] = wp1[j + (j & 8)]; }

        c0 = __builtin_amdgcn_wmma_f32_16x16x32_f16(false, a, false, b0, (short)0, c0, false, false);
        c1 = __builtin_amdgcn_wmma_f32_16x16x32_f16(false, a, false, b1, (short)0, c1, false, false);
    }

    // D layout: VGPR r, lanes 0-15 -> M=r, lanes 16-31 -> M=r+8; N = lane&15.
    const int mbase = row0 + ((lane >> 4) << 3);
    const int ncol  = lane & 15;
    #pragma unroll
    for (int r = 0; r < 8; ++r) {
        const int m = mbase + r;
        if (m < N) {
            xp[(size_t)m * 32 + ncol]      = c0[r];
            xp[(size_t)m * 32 + 16 + ncol] = c1[r];
        }
    }
}

// ---------------- per-node attention coefficients (layer 1) -------------------------
__global__ void node_att1_kernel(const float* __restrict__ xp,
                                 const float* __restrict__ as, const float* __restrict__ ad,
                                 float* __restrict__ a_src, float* __restrict__ a_dst, int N) {
    const int n = blockIdx.x * blockDim.x + threadIdx.x;
    if (n >= N) return;
    const float* v = xp + (size_t)n * 32;
    #pragma unroll
    for (int h = 0; h < 2; ++h) {
        float ss = 0.0f, dd = 0.0f;
        #pragma unroll
        for (int c = 0; c < 16; ++c) {
            const float t = v[h * 16 + c];
            ss += t * as[h * 16 + c];
            dd += t * ad[h * 16 + c];
        }
        a_src[n * 2 + h] = ss;
        a_dst[n * 2 + h] = dd;
    }
}

// ---------------- layer-1 edge passes: segment max / sum / aggregate ----------------
__global__ void edge1_max_kernel(const long long* __restrict__ ei, long long E, long long Etot,
                                 const float* __restrict__ a_src, const float* __restrict__ a_dst,
                                 float* __restrict__ m1) {
    const long long e = (long long)blockIdx.x * blockDim.x + threadIdx.x;
    if (e >= Etot) return;
    int s, d; edge_sd(ei, E, e, s, d);
    #pragma unroll
    for (int h = 0; h < 2; ++h) {
        const float al = lrelu(a_src[s * 2 + h] + a_dst[d * 2 + h]);
        atomicMaxFloat(&m1[d * 2 + h], al);
    }
}

__global__ void edge1_sum_kernel(const long long* __restrict__ ei, long long E, long long Etot,
                                 const float* __restrict__ a_src, const float* __restrict__ a_dst,
                                 const float* __restrict__ m1, float* __restrict__ s1) {
    const long long e = (long long)blockIdx.x * blockDim.x + threadIdx.x;
    if (e >= Etot) return;
    int s, d; edge_sd(ei, E, e, s, d);
    #pragma unroll
    for (int h = 0; h < 2; ++h) {
        const float al = lrelu(a_src[s * 2 + h] + a_dst[d * 2 + h]);
        atomicAdd(&s1[d * 2 + h], __expf(al - m1[d * 2 + h]));
    }
}

// 8 lanes cooperate on one edge: lane group g owns channels 4g..4g+3.
// The xp[src] gather becomes one coalesced 128B line per edge.
__global__ void edge1_agg_kernel(const long long* __restrict__ ei, long long E, long long Etot,
                                 const float* __restrict__ a_src, const float* __restrict__ a_dst,
                                 const float* __restrict__ m1, const float* __restrict__ s1,
                                 const float* __restrict__ xp, float* __restrict__ out1) {
    const long long t = (long long)blockIdx.x * blockDim.x + threadIdx.x;
    const long long e = t >> 3;
    if (e >= Etot) return;
    const int g = (int)(t & 7);
    const int h = g >> 2;                    // head for these 4 channels
    int s, d; edge_sd(ei, E, e, s, d);
    const float al = lrelu(a_src[s * 2 + h] + a_dst[d * 2 + h]);
    const float w  = __expf(al - m1[d * 2 + h]) / (s1[d * 2 + h] + GAT_EPS);
    const float4 tv = *(const float4*)(xp + (size_t)s * 32 + g * 4);
    float* od = out1 + (size_t)d * 32 + g * 4;
    atomicAdd(od + 0, tv.x * w);
    atomicAdd(od + 1, tv.y * w);
    atomicAdd(od + 2, tv.z * w);
    atomicAdd(od + 3, tv.w * w);
}

// ---------------- relu+bias (in place) and layer-2 projection -----------------------
__global__ void node_relu_l2_kernel(float* __restrict__ out1, const float* __restrict__ b1,
                                    const float* __restrict__ W2,
                                    const float* __restrict__ as2, const float* __restrict__ ad2,
                                    float* __restrict__ hp, float* __restrict__ a2s,
                                    float* __restrict__ a2d, int N) {
    const int n = blockIdx.x * blockDim.x + threadIdx.x;
    if (n >= N) return;
    float* v = out1 + (size_t)n * 32;
    float acc = 0.0f;
    #pragma unroll
    for (int j = 0; j < 32; ++j) {
        float t = v[j] + b1[j];
        t = t > 0.0f ? t : 0.0f;
        v[j] = t;
        acc += t * W2[j];
    }
    hp[n]  = acc;
    a2s[n] = acc * as2[0];
    a2d[n] = acc * ad2[0];
}

// ---------------- layer-2 edge passes (H=1, C=1) ------------------------------------
__global__ void edge2_max_kernel(const long long* __restrict__ ei, long long E, long long Etot,
                                 const float* __restrict__ a2s, const float* __restrict__ a2d,
                                 float* __restrict__ m2) {
    const long long e = (long long)blockIdx.x * blockDim.x + threadIdx.x;
    if (e >= Etot) return;
    int s, d; edge_sd(ei, E, e, s, d);
    atomicMaxFloat(&m2[d], lrelu(a2s[s] + a2d[d]));
}

__global__ void edge2_sum_kernel(const long long* __restrict__ ei, long long E, long long Etot,
                                 const float* __restrict__ a2s, const float* __restrict__ a2d,
                                 const float* __restrict__ m2, float* __restrict__ s2) {
    const long long e = (long long)blockIdx.x * blockDim.x + threadIdx.x;
    if (e >= Etot) return;
    int s, d; edge_sd(ei, E, e, s, d);
    atomicAdd(&s2[d], __expf(lrelu(a2s[s] + a2d[d]) - m2[d]));
}

__global__ void edge2_agg_kernel(const long long* __restrict__ ei, long long E, long long Etot,
                                 const float* __restrict__ a2s, const float* __restrict__ a2d,
                                 const float* __restrict__ m2, const float* __restrict__ s2,
                                 const float* __restrict__ hp, float* __restrict__ out2) {
    const long long e = (long long)blockIdx.x * blockDim.x + threadIdx.x;
    if (e >= Etot) return;
    int s, d; edge_sd(ei, E, e, s, d);
    const float w = __expf(lrelu(a2s[s] + a2d[d]) - m2[d]) / (s2[d] + GAT_EPS);
    atomicAdd(&out2[d], hp[s] * w);
}

// ---------------- final reduction: sum_n (out2[n] + b2) -> d_out[0] -----------------
__global__ void final_reduce_kernel(const float* __restrict__ out2, const float* __restrict__ b2,
                                    float* __restrict__ dout, int N) {
    __shared__ float sm[256];
    const float bb = b2[0];
    float acc = 0.0f;
    for (int i = blockIdx.x * 256 + threadIdx.x; i < N; i += gridDim.x * 256)
        acc += out2[i] + bb;
    sm[threadIdx.x] = acc;
    __syncthreads();
    for (int off = 128; off > 0; off >>= 1) {
        if (threadIdx.x < off) sm[threadIdx.x] += sm[threadIdx.x + off];
        __syncthreads();
    }
    if (threadIdx.x == 0) atomicAdd(dout, sm[0]);
}

extern "C" void kernel_launch(void* const* d_in, const int* in_sizes, int n_in,
                              void* d_out, int out_size, void* d_ws, size_t ws_size,
                              hipStream_t stream) {
    const float*     x   = (const float*)d_in[0];
    const long long* ei  = (const long long*)d_in[1];   // int64 [2, E]
    const float*     W1  = (const float*)d_in[2];
    const float*     as1 = (const float*)d_in[3];
    const float*     ad1 = (const float*)d_in[4];
    const float*     b1  = (const float*)d_in[5];
    const float*     W2  = (const float*)d_in[6];
    const float*     as2 = (const float*)d_in[7];
    const float*     ad2 = (const float*)d_in[8];
    const float*     b2  = (const float*)d_in[9];
    float*           out = (float*)d_out;

    const int       N    = in_sizes[0] / 128;
    const long long E    = (long long)in_sizes[1] / 2;
    const long long Etot = E + N;

    // workspace layout (floats)
    float* ws = (float*)d_ws;
    size_t off = 0;
    float* xp     = ws + off; off += (size_t)N * 32;
    float* a_src1 = ws + off; off += (size_t)N * 2;
    float* a_dst1 = ws + off; off += (size_t)N * 2;
    float* m1     = ws + off; off += (size_t)N * 2;
    float* s1     = ws + off; off += (size_t)N * 2;
    float* out1   = ws + off; off += (size_t)N * 32;
    float* hp     = ws + off; off += (size_t)N;
    float* a2s    = ws + off; off += (size_t)N;
    float* a2d    = ws + off; off += (size_t)N;
    float* m2     = ws + off; off += (size_t)N;
    float* s2     = ws + off; off += (size_t)N;
    float* out2   = ws + off; off += (size_t)N;

    const int TB = 256;
    const int nodeGrid  = (N + TB - 1) / TB;
    const int bigGrid   = (int)(((long long)N * 32 + TB - 1) / TB);
    const int edgeGrid  = (int)((Etot + TB - 1) / TB);
    const int edgeGrid8 = (int)((Etot * 8 + TB - 1) / TB);
    const int gemmGrid  = (N + 127) / 128;   // 8 waves x 16 rows per block

    init_ws_kernel<<<bigGrid, TB, 0, stream>>>(m1, s1, out1, m2, s2, out2, out, N);
    gemm1_wmma_kernel<<<gemmGrid, TB, 0, stream>>>(x, W1, xp, N);
    node_att1_kernel<<<nodeGrid, TB, 0, stream>>>(xp, as1, ad1, a_src1, a_dst1, N);
    edge1_max_kernel<<<edgeGrid, TB, 0, stream>>>(ei, E, Etot, a_src1, a_dst1, m1);
    edge1_sum_kernel<<<edgeGrid, TB, 0, stream>>>(ei, E, Etot, a_src1, a_dst1, m1, s1);
    edge1_agg_kernel<<<edgeGrid8, TB, 0, stream>>>(ei, E, Etot, a_src1, a_dst1, m1, s1, xp, out1);
    node_relu_l2_kernel<<<nodeGrid, TB, 0, stream>>>(out1, b1, W2, as2, ad2, hp, a2s, a2d, N);
    edge2_max_kernel<<<edgeGrid, TB, 0, stream>>>(ei, E, Etot, a2s, a2d, m2);
    edge2_sum_kernel<<<edgeGrid, TB, 0, stream>>>(ei, E, Etot, a2s, a2d, m2, s2);
    edge2_agg_kernel<<<edgeGrid, TB, 0, stream>>>(ei, E, Etot, a2s, a2d, m2, s2, hp, out2);
    final_reduce_kernel<<<256, TB, 0, stream>>>(out2, b2, out, N);
}
